// VisionEditAdaptor_12275016532570
// MI455X (gfx1250) — compile-verified, matrix operand order checked by reading, not run on hardware
//
#include <hip/hip_runtime.h>
#include <hip/hip_bf16.h>
#include <stdint.h>

// ---------------- problem constants ----------------
#define BATCH 8
#define LSEQ  1200
#define HID   4096
#define MID   1024
#define NHEAD 8
#define TN    576        // VE - VB
#define L2E   64
#define INFM  256
#define VB    35
#define MD    128        // MID / NHEAD
#define MROWS (BATCH*TN)   // 4608
#define EROWS (BATCH*L2E)  // 512

// ---------------- bf16 helpers ----------------
typedef __attribute__((ext_vector_type(16))) __bf16 v16bf;
typedef __attribute__((ext_vector_type(8)))  float  v8f;

__device__ __forceinline__ unsigned short f2us(float f) {
  union { float f; unsigned int u; } c; c.f = f;
  unsigned int u = c.u;
  unsigned int r = (u + 0x7FFFu + ((u >> 16) & 1u)) >> 16;  // RNE
  return (unsigned short)r;
}
__device__ __forceinline__ float us2f(unsigned short s) {
  union { unsigned int u; float f; } c; c.u = ((unsigned int)s) << 16;
  return c.f;
}

// ---------------- WMMA fragment loader ----------------
// 16-bit A (16x32) / Bt (16 rows x 32 K) fragment, row-major, ld = K.
// ISA layout: lanes 0-15 -> row (lane&15), K octet base = (lane>>4)*8;
// vector elem e: k = (e>>3)*16 + octet + (e&7).  Pairs are K-contiguous -> dword loads.
__device__ __forceinline__ v16bf load_frag(const unsigned short* __restrict__ base,
                                           int ld, int rbase, int k0, int lane) {
  const unsigned short* p =
      base + (size_t)(rbase + (lane & 15)) * ld + k0 + ((lane >> 4) << 3);
  v16bf r;
#pragma unroll
  for (int g = 0; g < 2; ++g) {
#pragma unroll
    for (int w = 0; w < 4; ++w) {
      union { unsigned int u; __bf16 h[2]; } c;
      c.u = *(const unsigned int*)(p + g * 16 + 2 * w);
      r[g * 8 + 2 * w]     = c.h[0];
      r[g * 8 + 2 * w + 1] = c.h[1];
    }
  }
  return r;
}

#define WMMA_BF16(A_, B_, C_) \
  __builtin_amdgcn_wmma_f32_16x16x32_bf16(false, A_, false, B_, (short)0, C_, false, false)

// ---------------- generic WMMA GEMM ----------------
// C[M,N] = A[M,K](bf16) * Bt[N,K](bf16)^T + bias, fp32 accum.
// Block: 256 thr = 8 waves (2x4), block tile 64(M) x 256(N); wave tile 32x64:
// 8 WMMAs per 6 fragment loads per 32-K step.
// __launch_bounds__(256,1): allow up to 512 VGPRs/wave -> no spills in the
// pipelined K-loop (Round-2 disasm showed scratch reloads at default budget).
// MODE 0: store fp32 (Cf) and/or bf16 (Cbf).
// MODE 1: fused epilogue  dst[b,VB+i,col] = src[b,VB+i,col] + v * gate[row]
template <int MODE>
__global__ __launch_bounds__(256, 1)
void gemm_bf16_wmma(const unsigned short* __restrict__ A,
                    const unsigned short* __restrict__ Bt,
                    const float* __restrict__ bias,
                    float* __restrict__ Cf,
                    unsigned short* __restrict__ Cbf,
                    const float* __restrict__ gate,
                    const float* __restrict__ srcFull,
                    float* __restrict__ dstFull,
                    int M, int N, int K) {
  const int lane = threadIdx.x & 31;
  const int wave = threadIdx.x >> 5;
  const int m0 = blockIdx.y * 64 + (wave >> 2) * 32;
  const int n0 = blockIdx.x * 256 + (wave & 3) * 64;

  v8f acc00 = {}, acc01 = {}, acc02 = {}, acc03 = {};
  v8f acc10 = {}, acc11 = {}, acc12 = {}, acc13 = {};

  for (int k0 = 0; k0 < K; k0 += 32) {
    v16bf a0 = load_frag(A, K, m0,      k0, lane);
    v16bf a1 = load_frag(A, K, m0 + 16, k0, lane);
    v16bf b0 = load_frag(Bt, K, n0,      k0, lane);
    v16bf b1 = load_frag(Bt, K, n0 + 16, k0, lane);
    v16bf b2 = load_frag(Bt, K, n0 + 32, k0, lane);
    v16bf b3 = load_frag(Bt, K, n0 + 48, k0, lane);
    acc00 = WMMA_BF16(a0, b0, acc00);
    acc01 = WMMA_BF16(a0, b1, acc01);
    acc02 = WMMA_BF16(a0, b2, acc02);
    acc03 = WMMA_BF16(a0, b3, acc03);
    acc10 = WMMA_BF16(a1, b0, acc10);
    acc11 = WMMA_BF16(a1, b1, acc11);
    acc12 = WMMA_BF16(a1, b2, acc12);
    acc13 = WMMA_BF16(a1, b3, acc13);
  }

  const int cn = lane & 15;
  const int rsel = (lane >> 4) << 3;  // C layout: vgpr r -> M=r (+8 for hi lanes)
#pragma unroll
  for (int t = 0; t < 8; ++t) {
    const v8f& acc = (t == 0) ? acc00 : (t == 1) ? acc01 : (t == 2) ? acc02 :
                     (t == 3) ? acc03 : (t == 4) ? acc10 : (t == 5) ? acc11 :
                     (t == 6) ? acc12 : acc13;
    const int mt = m0 + (t >> 2) * 16;
    const int nt = n0 + (t & 3) * 16;
#pragma unroll
    for (int r = 0; r < 8; ++r) {
      const int row = mt + rsel + r;
      const int col = nt + cn;
      float vv = acc[r] + bias[col];
      if (MODE == 0) {
        if (Cf)  Cf[(size_t)row * N + col] = vv;
        if (Cbf) Cbf[(size_t)row * N + col] = f2us(vv);
      } else {
        const int bb = row / TN;
        const int ii = row - bb * TN;
        const size_t idx = ((size_t)(bb * LSEQ + VB + ii)) * HID + col;
        dstFull[idx] = srcFull[idx] + vv * gate[row];
      }
    }
  }
}

// ---------------- full-tensor copy ----------------
__global__ void copy_f4(const float4* __restrict__ in, float4* __restrict__ out, int n) {
  int i = blockIdx.x * 256 + threadIdx.x;
  if (i < n) out[i] = in[i];
}

// ---------------- fp32 -> bf16 transpose (weights to N x K) ----------------
__global__ void transpose_f2bf(const float* __restrict__ in,
                               unsigned short* __restrict__ out, int R, int C) {
  __shared__ float tile[32][33];
  const int c0 = blockIdx.x * 32, r0 = blockIdx.y * 32;
  const int tx = threadIdx.x & 31, ty = threadIdx.x >> 5;  // 32x8
#pragma unroll
  for (int k = 0; k < 4; ++k)
    tile[ty + 8 * k][tx] = in[(size_t)(r0 + ty + 8 * k) * C + c0 + tx];
  __syncthreads();
#pragma unroll
  for (int k = 0; k < 4; ++k)
    out[(size_t)(c0 + ty + 8 * k) * R + r0 + tx] = f2us(tile[tx][ty + 8 * k]);
}

// ---------------- block reduction helper ----------------
__device__ __forceinline__ void block_reduce2(float& s, float& q,
                                              float* reds, float* redq) {
  const int tid = threadIdx.x;
  reds[tid] = s; redq[tid] = q;
  __syncthreads();
  for (int st = 128; st; st >>= 1) {
    if (tid < st) { reds[tid] += reds[tid + st]; redq[tid] += redq[tid + st]; }
    __syncthreads();
  }
  s = reds[0]; q = redq[0];
}

// ---------------- dual LayerNorm of img rows -> two bf16 matrices ----------------
__global__ void ln_img_dual(const float* __restrict__ lo,
                            const float* __restrict__ g1, const float* __restrict__ b1,
                            const float* __restrict__ g2, const float* __restrict__ b2,
                            unsigned short* __restrict__ o1, unsigned short* __restrict__ o2) {
  __shared__ float reds[256], redq[256];
  const int row = blockIdx.x;                    // 0..4607
  const int b = row / TN, i = row - b * TN;
  const float* src = lo + ((size_t)(b * LSEQ + VB + i)) * HID;
  float s = 0.f, q = 0.f;
  for (int d = threadIdx.x; d < HID; d += 256) { float x = src[d]; s += x; q += x * x; }
  block_reduce2(s, q, reds, redq);
  const float mean = s * (1.0f / HID);
  const float inv = rsqrtf(q * (1.0f / HID) - mean * mean + 1e-5f);
  for (int d = threadIdx.x; d < HID; d += 256) {
    float xn = (src[d] - mean) * inv;
    o1[(size_t)row * HID + d] = f2us(xn * g1[d] + b1[d]);
    o2[(size_t)row * HID + d] = f2us(xn * g2[d] + b2[d]);
  }
}

// ---------------- LayerNorm of edit_reps rows -> bf16 ----------------
__global__ void ln_edit_rows(const float* __restrict__ e,
                             const float* __restrict__ g, const float* __restrict__ bb,
                             unsigned short* __restrict__ o) {
  __shared__ float reds[256], redq[256];
  const int row = blockIdx.x;                    // 0..511
  const float* src = e + (size_t)row * HID;
  float s = 0.f, q = 0.f;
  for (int d = threadIdx.x; d < HID; d += 256) { float x = src[d]; s += x; q += x * x; }
  block_reduce2(s, q, reds, redq);
  const float mean = s * (1.0f / HID);
  const float inv = rsqrtf(q * (1.0f / HID) - mean * mean + 1e-5f);
  for (int d = threadIdx.x; d < HID; d += 256)
    o[(size_t)row * HID + d] = f2us(((src[d] - mean) * inv) * g[d] + bb[d]);
}

// ---------------- pe = resmap(ln(edit_reps[b, prompt_end[b]])) (8 rows) ----------------
__global__ void pe_kernel(const float* __restrict__ edit, const int* __restrict__ pend,
                          const float* __restrict__ g, const float* __restrict__ bb,
                          const float* __restrict__ Wi, const float* __restrict__ bi,
                          const float* __restrict__ Wo, const float* __restrict__ bo,
                          float* __restrict__ pe) {
  __shared__ float lnv[HID];
  __shared__ float hsh[INFM];
  __shared__ float reds[256], redq[256];
  const int b = blockIdx.x;
  const int tid = threadIdx.x;
  const float* src = edit + ((size_t)b * L2E + pend[b]) * HID;
  float s = 0.f, q = 0.f;
  for (int d = tid; d < HID; d += 256) { float x = src[d]; s += x; q += x * x; }
  block_reduce2(s, q, reds, redq);
  const float mean = s * (1.0f / HID);
  const float inv = rsqrtf(q * (1.0f / HID) - mean * mean + 1e-5f);
  for (int d = tid; d < HID; d += 256) lnv[d] = (src[d] - mean) * inv * g[d] + bb[d];
  __syncthreads();
  float acc = 0.f;
  for (int d = 0; d < HID; ++d) acc += lnv[d] * Wi[(size_t)d * INFM + tid];
  hsh[tid] = fmaxf(acc + bi[tid], 0.0f);
  __syncthreads();
  for (int d = tid; d < HID; d += 256) {
    float a = 0.f;
    for (int j = 0; j < INFM; ++j) a += hsh[j] * Wo[(size_t)j * HID + d];
    pe[(size_t)b * HID + d] = a + bo[d] + lnv[d];
  }
}

// ---------------- wpe[b,j] = sum_d Wo[j,d] * pe[b,d]  (folds Wo GEMM away) ----
__global__ void wpe_kernel(const float* __restrict__ Wo, const float* __restrict__ pe,
                           float* __restrict__ wpe) {
  const int b = blockIdx.x, j = threadIdx.x;
  const float* wo = Wo + (size_t)j * HID;
  const float* p = pe + (size_t)b * HID;
  float a = 0.f;
  for (int d = 0; d < HID; ++d) a += wo[d] * p[d];
  wpe[b * INFM + j] = a;
}

// ---------------- gate: inf[row] = sigmoid((relu(h1).wpe + (ln+bo).pe)/(H*sqrt(hd)))
__global__ void gate_finalize(const float* __restrict__ h1, const float* __restrict__ wpe,
                              const unsigned short* __restrict__ imln,
                              const float* __restrict__ bo, const float* __restrict__ pe,
                              float* __restrict__ inf) {
  __shared__ float reds[256], redq[256];
  const int row = blockIdx.x;
  const int b = row / TN;
  const int tid = threadIdx.x;
  float part = fmaxf(h1[(size_t)row * INFM + tid], 0.0f) * wpe[b * INFM + tid];
  const unsigned short* li = imln + (size_t)row * HID;
  const float* p = pe + (size_t)b * HID;
  for (int d = tid; d < HID; d += 256) part += (us2f(li[d]) + bo[d]) * p[d];
  float dummy = 0.f;
  block_reduce2(part, dummy, reds, redq);
  if (tid == 0)
    inf[row] = 1.0f / (1.0f + __expf(-part * (1.0f / (8.0f * 22.627416997969522f))));
}

// ---------------- attention: per (b,h), 64 keys, 576 queries ----------------
// K tile is DMA'd into LDS with global_load_async_to_lds_b128 (ASYNCcnt path);
// V tile goes through VGPRs because it is converted to bf16 in flight.
__global__ __launch_bounds__(256, 1)
void attn_kernel(const float* __restrict__ q, const float* __restrict__ k,
                 const float* __restrict__ v, const float* __restrict__ mask,
                 unsigned short* __restrict__ obf) {
  __shared__ float ks[L2E][MD];           // 32 KB (async DMA target)
  __shared__ unsigned short vs[L2E][MD];  // 16 KB
  const int bh = blockIdx.x;
  const int b = bh >> 3, h = bh & 7;

  // ---- async K-tile load: 2048 float4s, 256 threads x 8 issues ----
  const unsigned ksBase = (unsigned)(uintptr_t)(&ks[0][0]);  // low 32 bits = LDS offset
#pragma unroll
  for (int i = 0; i < 8; ++i) {
    const int e  = threadIdx.x + 256 * i;   // float4 index in 64x128 tile
    const int u  = e >> 5;                  // key row
    const int d4 = e & 31;                  // float4 within row
    const float* g = k + (size_t)(b * L2E + u) * MID + h * MD + d4 * 4;
    const unsigned lds = ksBase + (unsigned)((u * MD + d4 * 4) * 4);
    asm volatile("global_load_async_to_lds_b128 %0, %1, off"
                 :: "v"(lds), "v"(g) : "memory");
  }
  // ---- V tile with fp32->bf16 conversion ----
  for (int t = threadIdx.x; t < L2E * MD; t += 256) {
    const int u = t >> 7, d = t & 127;
    vs[u][d] = f2us(v[(size_t)(b * L2E + u) * MID + h * MD + d]);
  }
  asm volatile("s_wait_asynccnt 0" ::: "memory");
  __syncthreads();

  const int lane = threadIdx.x & 31;
  const int wave = threadIdx.x >> 5;
  const float m0 = (mask[b * L2E + lane] - 1.0f) * 9999999999.0f;
  const float m1 = (mask[b * L2E + 32 + lane] - 1.0f) * 9999999999.0f;
  const float scale = 0.08838834764831845f;  // 1/sqrt(128)
  for (int l = wave; l < TN; l += 8) {
    const int row = b * TN + l;
    const float4 qv = *(const float4*)(q + (size_t)row * MID + h * MD + lane * 4);
    float s0 = 0.f, s1 = 0.f;
    for (int u = 0; u < L2E; ++u) {
      float part = qv.x * ks[u][lane * 4] + qv.y * ks[u][lane * 4 + 1] +
                   qv.z * ks[u][lane * 4 + 2] + qv.w * ks[u][lane * 4 + 3];
      for (int off = 16; off; off >>= 1) part += __shfl_xor(part, off, 32);
      if (lane == (u & 31)) { if (u < 32) s0 = part; else s1 = part; }
    }
    s0 = s0 * scale + m0;
    s1 = s1 * scale + m1;
    float mx = fmaxf(s0, s1);
    for (int off = 16; off; off >>= 1) mx = fmaxf(mx, __shfl_xor(mx, off, 32));
    const float e0 = __expf(s0 - mx), e1 = __expf(s1 - mx);
    float sum = e0 + e1;
    for (int off = 16; off; off >>= 1) sum += __shfl_xor(sum, off, 32);
    const float rs = 1.0f / sum;
    const float p0 = e0 * rs, p1 = e1 * rs;
    float o0 = 0.f, o1 = 0.f, o2 = 0.f, o3 = 0.f;
    for (int u = 0; u < L2E; ++u) {
      const float pu = __shfl(u < 32 ? p0 : p1, u & 31, 32);
      o0 += pu * us2f(vs[u][lane * 4]);
      o1 += pu * us2f(vs[u][lane * 4 + 1]);
      o2 += pu * us2f(vs[u][lane * 4 + 2]);
      o3 += pu * us2f(vs[u][lane * 4 + 3]);
    }
    const size_t ob = (size_t)row * MID + h * MD + lane * 4;
    obf[ob] = f2us(o0); obf[ob + 1] = f2us(o1);
    obf[ob + 2] = f2us(o2); obf[ob + 3] = f2us(o3);
  }
}

// =====================================================================
extern "C" void kernel_launch(void* const* d_in, const int* in_sizes, int n_in,
                              void* d_out, int out_size, void* d_ws, size_t ws_size,
                              hipStream_t stream) {
  const float* layer_outpt = (const float*)d_in[0];
  const float* edit_reps   = (const float*)d_in[1];
  const float* att_mask    = (const float*)d_in[2];
  const int*   prompt_end  = (const int*)d_in[3];
  const float* mlp_begin_W = (const float*)d_in[6];
  const float* mlp_begin_b = (const float*)d_in[7];
  const float* q_W = (const float*)d_in[8];
  const float* q_b = (const float*)d_in[9];
  const float* k_W = (const float*)d_in[10];
  const float* k_b = (const float*)d_in[11];
  const float* v_W = (const float*)d_in[12];
  const float* v_b = (const float*)d_in[13];
  const float* mlp_end_W = (const float*)d_in[14];
  const float* mlp_end_b = (const float*)d_in[15];
  const float* ln_img_g = (const float*)d_in[16];
  const float* ln_img_b = (const float*)d_in[17];
  const float* ln_edit_g = (const float*)d_in[18];
  const float* ln_edit_b = (const float*)d_in[19];
  const float* im_ln_img_g = (const float*)d_in[20];
  const float* im_ln_img_b = (const float*)d_in[21];
  const float* im_ln_edit_g = (const float*)d_in[22];
  const float* im_ln_edit_b = (const float*)d_in[23];
  const float* im_img_Wi = (const float*)d_in[24];
  const float* im_img_bi = (const float*)d_in[25];
  const float* im_img_Wo = (const float*)d_in[26];
  const float* im_img_bo = (const float*)d_in[27];
  const float* im_tok_Wi = (const float*)d_in[28];
  const float* im_tok_bi = (const float*)d_in[29];
  const float* im_tok_Wo = (const float*)d_in[30];
  const float* im_tok_bo = (const float*)d_in[31];
  float* out = (float*)d_out;

  // ---- workspace carve-out (all L2-resident: ~165 MB vs 192 MB L2) ----
  char* W = (char*)d_ws;
  size_t off = 0;
  auto take = [&](size_t bytes) -> void* {
    void* p = W + off;
    off += (bytes + 255) & ~(size_t)255;
    return p;
  };
  unsigned short* lnimg = (unsigned short*)take((size_t)MROWS * HID * 2);
  unsigned short* imln  = (unsigned short*)take((size_t)MROWS * HID * 2);
  unsigned short* nebf  = (unsigned short*)take((size_t)EROWS * HID * 2);
  unsigned short* xbf   = (unsigned short*)take((size_t)MROWS * MID * 2);
  unsigned short* obf   = (unsigned short*)take((size_t)MROWS * MID * 2);
  float* h1f  = (float*)take((size_t)MROWS * INFM * 4);
  float* qf   = (float*)take((size_t)MROWS * MID * 4);
  float* kf   = (float*)take((size_t)EROWS * MID * 4);
  float* vf   = (float*)take((size_t)EROWS * MID * 4);
  float* pef  = (float*)take((size_t)BATCH * HID * 4);
  float* wpef = (float*)take((size_t)BATCH * INFM * 4);
  float* inff = (float*)take((size_t)MROWS * 4);
  unsigned short* beginT = (unsigned short*)take((size_t)HID * MID * 2);
  unsigned short* qT     = (unsigned short*)take((size_t)MID * MID * 2);
  unsigned short* kT     = (unsigned short*)take((size_t)HID * MID * 2);
  unsigned short* vT     = (unsigned short*)take((size_t)HID * MID * 2);
  unsigned short* endT   = (unsigned short*)take((size_t)MID * HID * 2);
  unsigned short* imWiT  = (unsigned short*)take((size_t)HID * INFM * 2);

  // 1) out <- layer_outpt (slice overwritten later by fused epilogue)
  {
    const int n4 = BATCH * LSEQ * HID / 4;
    copy_f4<<<(n4 + 255) / 256, 256, 0, stream>>>((const float4*)layer_outpt,
                                                  (float4*)out, n4);
  }
  // 2) weight transposes -> bf16 [N x K]
  transpose_f2bf<<<dim3(MID / 32, HID / 32), 256, 0, stream>>>(mlp_begin_W, beginT, HID, MID);
  transpose_f2bf<<<dim3(MID / 32, MID / 32), 256, 0, stream>>>(q_W, qT, MID, MID);
  transpose_f2bf<<<dim3(MID / 32, HID / 32), 256, 0, stream>>>(k_W, kT, HID, MID);
  transpose_f2bf<<<dim3(MID / 32, HID / 32), 256, 0, stream>>>(v_W, vT, HID, MID);
  transpose_f2bf<<<dim3(HID / 32, MID / 32), 256, 0, stream>>>(mlp_end_W, endT, MID, HID);
  transpose_f2bf<<<dim3(INFM / 32, HID / 32), 256, 0, stream>>>(im_img_Wi, imWiT, HID, INFM);

  // 3) LayerNorms
  ln_img_dual<<<MROWS, 256, 0, stream>>>(layer_outpt, ln_img_g, ln_img_b,
                                         im_ln_img_g, im_ln_img_b, lnimg, imln);
  ln_edit_rows<<<EROWS, 256, 0, stream>>>(edit_reps, ln_edit_g, ln_edit_b, nebf);

  // 4) pe / wpe (gate small path; Wo GEMM algebraically folded into wpe)
  pe_kernel<<<BATCH, 256, 0, stream>>>(edit_reps, prompt_end, im_ln_edit_g, im_ln_edit_b,
                                       im_tok_Wi, im_tok_bi, im_tok_Wo, im_tok_bo, pef);
  wpe_kernel<<<BATCH, 256, 0, stream>>>(im_img_Wo, pef, wpef);

  // 5) gate GEMM: h1 = im_ln(img) @ im_img_Wi + bi   [4608 x 256, K=4096]
  gemm_bf16_wmma<0><<<dim3(INFM / 256, MROWS / 64), 256, 0, stream>>>(
      imln, imWiT, im_img_bi, h1f, nullptr, nullptr, nullptr, nullptr, MROWS, INFM, HID);
  gate_finalize<<<MROWS, 256, 0, stream>>>(h1f, wpef, imln, im_img_bo, pef, inff);

  // 6) x = ln(img) @ mlp_begin + b (store bf16 for q GEMM)
  gemm_bf16_wmma<0><<<dim3(MID / 256, MROWS / 64), 256, 0, stream>>>(
      lnimg, beginT, mlp_begin_b, nullptr, xbf, nullptr, nullptr, nullptr, MROWS, MID, HID);
  // 7) q = x @ q_W + q_b
  gemm_bf16_wmma<0><<<dim3(MID / 256, MROWS / 64), 256, 0, stream>>>(
      xbf, qT, q_b, qf, nullptr, nullptr, nullptr, nullptr, MROWS, MID, MID);
  // 8) k, v = ln(edit) @ {k_W, v_W} + b
  gemm_bf16_wmma<0><<<dim3(MID / 256, EROWS / 64), 256, 0, stream>>>(
      nebf, kT, k_b, kf, nullptr, nullptr, nullptr, nullptr, EROWS, MID, HID);
  gemm_bf16_wmma<0><<<dim3(MID / 256, EROWS / 64), 256, 0, stream>>>(
      nebf, vT, v_b, vf, nullptr, nullptr, nullptr, nullptr, EROWS, MID, HID);

  // 9) attention (64 keys) -> o (bf16)
  attn_kernel<<<BATCH * NHEAD, 256, 0, stream>>>(qf, kf, vf, att_mask, obf);

  // 10) fused: out[:, VB:VE] = img + (o @ mlp_end + b) * inf
  gemm_bf16_wmma<1><<<dim3(HID / 256, MROWS / 64), 256, 0, stream>>>(
      obf, endT, mlp_end_b, nullptr, nullptr, inff, layer_outpt, out, MROWS, HID, MID);

  (void)in_sizes; (void)n_in; (void)out_size; (void)ws_size;
}
